// VectorQuant_68015102100091
// MI455X (gfx1250) — matrix operand order repaired and checked by reading, not compiled, not run on hardware
//
#include <hip/hip_runtime.h>
#include <hip/hip_bf16.h>
#include <math.h>

typedef __bf16 v16bf __attribute__((ext_vector_type(16)));
typedef float  v8f   __attribute__((ext_vector_type(8)));

#define BS_TOTAL  131072          // 16*8192 rows
#define LVEC      128             // vector length
#define KC        512             // codebook size
#define NT        (KC / 16)       // 32 N-tiles of 16 codes
#define ROW_TILES (BS_TOTAL / 16) // 8192 row tiles

// d_out layout (floats): out0 | out1 | out2 | entropy
#define OUT0_OFF 0
#define OUT1_OFF 16777216
#define OUT2_OFF 16908288
#define ENT_OFF  17039360

// workspace layout (bytes): ebh | ebl | e2 | hist
#define WS_EBH  0
#define WS_EBL  131072
#define WS_E2   262144
#define WS_HIST 264192

// LDS layout (bytes): frag hi | frag lo | e2 | per-wave idx scratch
#define LDS_COPY_BYTES 264192      // ebh+ebl+e2, contiguous in ws
#define LDS_IDX_OFF    264192
#define LDS_TOTAL      264704

// ---------------------------------------------------------------------------
// Pack embedding into WMMA B-fragment layout, split f32 -> bf16 hi + lo.
// B matrix (32x16, KxN) per 16x16x32 step: lane ln (n = ln&15, h = ln>>4)
// holds 16 contiguous bf16: E[16t + n, 32q + 16h + e], e = 0..15.
// ---------------------------------------------------------------------------
__global__ void vq_pack_emb(const float* __restrict__ emb,
                            unsigned short* __restrict__ ebh,
                            unsigned short* __restrict__ ebl) {
    int gid = blockIdx.x * 256 + threadIdx.x;    // 0 .. 65535
    int n = gid >> 7;                            // code index
    int l = gid & 127;                           // element index
    float v = emb[gid];
    unsigned u = __float_as_uint(v);
    unsigned short hi = (unsigned short)(u >> 16);
    float hif = __uint_as_float(u & 0xFFFF0000u);
    float lof = v - hif;                         // exact residual
    unsigned short lo = (unsigned short)(__float_as_uint(lof) >> 16);

    int t = n >> 4, nn = n & 15;
    int q = l >> 5, lr = l & 31;
    int h = lr >> 4, e = lr & 15;
    int ln = h * 16 + nn;
    int dst = (((t * 4 + q) * 32) + ln) * 16 + e;
    ebh[dst] = hi;
    ebl[dst] = lo;
}

// e2[k] = ||E[k]||^2 (full f32), and zero the histogram each launch.
__global__ void vq_e2_hist(const float* __restrict__ emb,
                           float* __restrict__ e2,
                           unsigned* __restrict__ hist) {
    int n = blockIdx.x * 256 + threadIdx.x;
    if (n < KC) {
        const float* row = emb + n * LVEC;
        float s = 0.0f;
        #pragma unroll 4
        for (int l = 0; l < LVEC; ++l) s = fmaf(row[l], row[l], s);
        e2[n] = s;
        hist[n] = 0u;
    }
}

// ---------------------------------------------------------------------------
// Main fused kernel: GEMM (split-bf16 WMMA) + argmin + gather + residuals.
// One wave owns a 16-row tile; codebook fragments live in LDS (258 KB).
// Three independent accumulator chains (hi*ehi, lo*ehi, hi*elo) keep the
// matrix pipeline free of back-to-back RAW dependences.
// ---------------------------------------------------------------------------
__global__ __launch_bounds__(256) void vq_main(
        const float* __restrict__ x, const float* __restrict__ emb,
        const uint4* __restrict__ ws_base, unsigned* __restrict__ hist,
        float* __restrict__ out0, float* __restrict__ out1,
        float* __restrict__ out2) {
    extern __shared__ uint4 smem4[];
    // Stage packed codebook (hi||lo||e2) into LDS, 16 B per op.
    {
        uint4* dst = smem4;
        for (int i = threadIdx.x; i < LDS_COPY_BYTES / 16; i += 256)
            dst[i] = ws_base[i];
    }
    __syncthreads();

    char* smem = (char*)smem4;
    const unsigned short* lbh = (const unsigned short*)smem;             // hi frags
    const unsigned short* lbl = (const unsigned short*)(smem + WS_EBL);  // lo frags
    const float* le2 = (const float*)(smem + WS_E2);
    int* lidx = (int*)(smem + LDS_IDX_OFF);

    const int ln = threadIdx.x & 31;
    const int wv = threadIdx.x >> 5;
    const int m  = ln & 15;   // A-row / C-column slot
    const int h  = ln >> 4;   // K/M half select

    for (int tile = blockIdx.x * 8 + wv; tile < ROW_TILES; tile += gridDim.x * 8) {
        const int row0 = tile * 16;
        const float* xrow = x + (size_t)(row0 + m) * LVEC;

        // Build A fragments (16x32 bf16 layout, ISA 7.12.2): lane m=ln&15,
        // elements 0..15 -> cols 32q+16h+e (contiguous, matches B packing).
        v16bf ahi[4], alo[4];
        #pragma unroll
        for (int q = 0; q < 4; ++q) {
            union { v16bf v; unsigned short s[16]; } uh, ul;
            const int base1 = q * 32 + h * 16;
            const int base2 = q * 32 + h * 16 + 8;
            #pragma unroll
            for (int e = 0; e < 8; ++e) {
                float v1 = xrow[base1 + e];
                float v2 = xrow[base2 + e];
                unsigned u1 = __float_as_uint(v1);
                unsigned u2 = __float_as_uint(v2);
                uh.s[e]     = (unsigned short)(u1 >> 16);
                uh.s[e + 8] = (unsigned short)(u2 >> 16);
                float h1 = __uint_as_float(u1 & 0xFFFF0000u);
                float h2 = __uint_as_float(u2 & 0xFFFF0000u);
                ul.s[e]     = (unsigned short)(__float_as_uint(v1 - h1) >> 16);
                ul.s[e + 8] = (unsigned short)(__float_as_uint(v2 - h2) >> 16);
            }
            ahi[q] = uh.v;
            alo[q] = ul.v;
        }

        float best[8];
        int   bidx[8];
        #pragma unroll
        for (int r = 0; r < 8; ++r) { best[r] = 3.4e38f; bidx[r] = 0; }

        // Sweep the 32 N-tiles of the codebook.
        for (int t = 0; t < NT; ++t) {
            const v8f z = {0.f, 0.f, 0.f, 0.f, 0.f, 0.f, 0.f, 0.f};
            v8f acc0 = z, acc1 = z, acc2 = z;   // hi*ehi, lo*ehi, hi*elo
            #pragma unroll
            for (int q = 0; q < 4; ++q) {
                const int fo = ((t * 4 + q) * 32 + ln) * 16;
                v16bf bh = *(const v16bf*)(lbh + fo);
                v16bf bl = *(const v16bf*)(lbl + fo);
                acc0 = __builtin_amdgcn_wmma_f32_16x16x32_bf16(
                           false, ahi[q], false, bh, (short)0, acc0, false, false);
                acc1 = __builtin_amdgcn_wmma_f32_16x16x32_bf16(
                           false, alo[q], false, bh, (short)0, acc1, false, false);
                acc2 = __builtin_amdgcn_wmma_f32_16x16x32_bf16(
                           false, ahi[q], false, bl, (short)0, acc2, false, false);
            }
            // C/D layout: VGPR r, lane -> row m' = r + 8h, col n = ln&15.
            const float e2v = le2[t * 16 + m];
            const int   code = t * 16 + m;
            #pragma unroll
            for (int r = 0; r < 8; ++r) {
                float sc = fmaf(-2.0f, acc0[r], e2v);   // ||e||^2 - 2 x.e
                sc = fmaf(-2.0f, acc1[r], sc);
                sc = fmaf(-2.0f, acc2[r], sc);
                if (sc < best[r]) { best[r] = sc; bidx[r] = code; }
            }
        }

        // Reduce argmin across the 16 N-lanes of each half (masks 1,2,4,8
        // stay inside a 16-lane group). Tie-break to the LOWER index
        // (jnp.argmin picks the first occurrence).
        #pragma unroll
        for (int r = 0; r < 8; ++r) {
            #pragma unroll
            for (int mask = 1; mask < 16; mask <<= 1) {
                float os = __shfl_xor(best[r], mask, 32);
                int   oi = __shfl_xor(bidx[r], mask, 32);
                if (os < best[r] || (os == best[r] && oi < bidx[r])) {
                    best[r] = os; bidx[r] = oi;
                }
            }
            if (m == 0) {   // lanes 0 and 16: rows r and r+8
                lidx[wv * 16 + r + h * 8] = bidx[r];
                atomicAdd(&hist[bidx[r]], 1u);
            }
        }
        // Same wave wrote lidx; wave-synchronous, no barrier needed.

        // Gather + residuals: whole wave per row, 4 floats per lane.
        #pragma unroll 1
        for (int mm = 0; mm < 16; ++mm) {
            const int idx = lidx[wv * 16 + mm];
            const float4* ev4 = (const float4*)(emb + (size_t)idx * LVEC);
            const float4* xv4 = (const float4*)(x + (size_t)(row0 + mm) * LVEC);
            float4 e4 = ev4[ln];
            float4 x4 = xv4[ln];
            // out0 = (output - x) + x, emulated exactly as in the reference
            float4 o4;
            o4.x = (e4.x - x4.x) + x4.x;
            o4.y = (e4.y - x4.y) + x4.y;
            o4.z = (e4.z - x4.z) + x4.z;
            o4.w = (e4.w - x4.w) + x4.w;
            ((float4*)out0)[(size_t)(row0 + mm) * 32 + ln] = o4;
            float dx = x4.x - e4.x, dy = x4.y - e4.y;
            float dz = x4.z - e4.z, dw = x4.w - e4.w;
            float d = dx * dx + dy * dy + dz * dz + dw * dw;
            #pragma unroll
            for (int mask = 16; mask >= 1; mask >>= 1)
                d += __shfl_xor(d, mask, 32);
            if (ln == 0) {
                out1[row0 + mm] = d;
                out2[row0 + mm] = d;   // + sum((x-x0)^2) == 0 since x == x0
            }
        }
    }
}

// Entropy of the code-usage histogram: -sum p log p, p = count / 131072.
__global__ void vq_entropy(const unsigned* __restrict__ hist,
                           float* __restrict__ outp) {
    __shared__ float red[256];
    const int k = threadIdx.x;
    float s = 0.0f;
    for (int b = k; b < KC; b += 256) {
        unsigned c = hist[b];
        if (c > 0u) {
            float p = (float)c * (1.0f / 131072.0f);
            s -= p * logf(p);
        }
    }
    red[k] = s;
    __syncthreads();
    for (int off = 128; off > 0; off >>= 1) {
        if (k < off) red[k] += red[k + off];
        __syncthreads();
    }
    if (k == 0) outp[0] = red[0];
}

extern "C" void kernel_launch(void* const* d_in, const int* in_sizes, int n_in,
                              void* d_out, int out_size, void* d_ws, size_t ws_size,
                              hipStream_t stream) {
    const float* x   = (const float*)d_in[0];   // (16,8192,1,128) f32
    const float* emb = (const float*)d_in[1];   // (1,512,128) f32
    float* out = (float*)d_out;
    char*  ws  = (char*)d_ws;

    unsigned short* ebh  = (unsigned short*)(ws + WS_EBH);
    unsigned short* ebl  = (unsigned short*)(ws + WS_EBL);
    float*          e2   = (float*)(ws + WS_E2);
    unsigned*       hist = (unsigned*)(ws + WS_HIST);

    vq_pack_emb<<<256, 256, 0, stream>>>(emb, ebh, ebl);
    vq_e2_hist<<<2, 256, 0, stream>>>(emb, e2, hist);
    vq_main<<<256, 256, LDS_TOTAL, stream>>>(x, emb, (const uint4*)ws, hist,
                                             out + OUT0_OFF, out + OUT1_OFF,
                                             out + OUT2_OFF);
    vq_entropy<<<1, 256, 0, stream>>>(hist, out + ENT_OFF);
}
